// EmbedLSTM_43465069035645
// MI455X (gfx1250) — matrix operand order repaired and checked by reading, not compiled, not run on hardware
//
#include <hip/hip_runtime.h>
#include <hip/hip_bf16.h>

typedef __attribute__((ext_vector_type(16))) __bf16 v16bf;
typedef __attribute__((ext_vector_type(8)))  float  v8f;
typedef __attribute__((ext_vector_type(4)))  unsigned v4u;
typedef __attribute__((ext_vector_type(8)))  int    v8i_;
typedef __attribute__((ext_vector_type(4)))  int    v4i_;

#define INP 2048
#define HID 2048
#define SEQ 8192
#define G4  (4 * HID)
#define NB  128      // persistent blocks, 1 per WGP (needs ~267KB of the WGP's 320KB LDS)
#define JB  16       // hidden indices per block

// LDS layout for the persistent kernel (dynamic shared):
//   [0, WLDS)          : W_hg slice, 64 rows (4 gates x 16 cols), padded row stride
//   [WLDS, +4KB)       : h broadcast buffer (TDM destination)
//   [.., +512B)        : gate partial-sum buffer
#define WROW        2056                     // 2048 + 8 pad -> lane bank offset = 4 banks
#define WLDS_BYTES  (64 * WROW * 2)          // 263168 B
#define HSH_OFF     WLDS_BYTES
#define GBUF_OFF    (WLDS_BYTES + HID * 2)
#define SMEM_BYTES  (GBUF_OFF + 2 * 4 * 16 * (int)sizeof(float))
static_assert(SMEM_BYTES <= 320 * 1024, "exceeds 320KB WGP LDS");

union ABf {
    v16bf v;
    uint4 q[2];
};

__device__ __forceinline__ __bf16 f2bf(float f) {
    unsigned u = __float_as_uint(f);
    unsigned short s = (unsigned short)((u + 0x7FFFu + ((u >> 16) & 1u)) >> 16);
    __bf16 b;
    __builtin_memcpy(&b, &s, 2);
    return b;
}

__device__ __forceinline__ float sigmoidf_(float x) {
    return 1.0f / (1.0f + expf(-x));
}

// ---------------- prep kernels ----------------

__global__ void k_cast_bf16(const float* __restrict__ src, __bf16* __restrict__ dst, int n) {
    int i = blockIdx.x * blockDim.x + threadIdx.x;
    if (i < n) dst[i] = f2bf(src[i]);
}

// src: [K, N] fp32 row-major  ->  dst: [N, K] bf16 row-major
__global__ void k_tcast(const float* __restrict__ src, __bf16* __restrict__ dst, int K, int N) {
    __shared__ float tile[32][33];
    const int kb = blockIdx.x * 32;
    const int nb = blockIdx.y * 32;
    const int tx = threadIdx.x & 31;
    const int ty = threadIdx.x >> 5;
    for (int r = ty; r < 32; r += 8)
        tile[r][tx] = src[(size_t)(kb + r) * N + nb + tx];
    __syncthreads();
    for (int r = ty; r < 32; r += 8)
        dst[(size_t)(nb + r) * K + kb + tx] = f2bf(tile[tx][r]);
}

__global__ void k_init_state(__bf16* __restrict__ h0, __bf16* __restrict__ h1,
                             unsigned* __restrict__ bar) {
    int i = blockIdx.x * blockDim.x + threadIdx.x;
    if (i < HID) {
        h0[i] = f2bf(0.0f);
        h1[i] = f2bf(0.0f);
    }
    if (i < 2) bar[i] = 0u;   // bar[0]=arrive counter, bar[1]=generation
}

// ---------------- phase 1: xg = x @ W_xg + b ----------------
__global__ void k_gemm_xw(const __bf16* __restrict__ xb,
                          const __bf16* __restrict__ wT,
                          const float*  __restrict__ bias,
                          float* __restrict__ xg) {
    const int lane = threadIdx.x & 31;
    const int wid  = blockIdx.x * (blockDim.x >> 5) + (threadIdx.x >> 5);
    const int nTiles = G4 / 64;
    const int mT = wid / nTiles;
    const int nT = wid % nTiles;
    const int row0 = mT * 16;
    const int col0 = nT * 64;
    const bool hi = lane >= 16;
    const int  l15 = lane & 15;

    const __bf16* aRow = xb + (size_t)(row0 + l15) * INP;

    v8f acc[4] = {};
    for (int k0 = 0; k0 < INP; k0 += 32) {
        ABf a;
        const int ka = k0 + (hi ? 16 : 0);
        a.q[0] = *(const uint4*)(aRow + ka);
        a.q[1] = *(const uint4*)(aRow + ka + 8);
        __builtin_prefetch(aRow + k0 + 256, 0, 1);
#pragma unroll
        for (int s = 0; s < 4; ++s) {
            const __bf16* bRow = wT + (size_t)(col0 + s * 16 + l15) * INP + k0 + (hi ? 16 : 0);
            ABf b;
            b.q[0] = ((const uint4*)bRow)[0];
            b.q[1] = ((const uint4*)bRow)[1];
            acc[s] = __builtin_amdgcn_wmma_f32_16x16x32_bf16(
                false, a.v, false, b.v, (short)0, acc[s], false, false);
        }
    }
    const int mBase = row0 + (hi ? 8 : 0);
#pragma unroll
    for (int s = 0; s < 4; ++s) {
        const int col = col0 + s * 16 + l15;
        const float bv = bias[col];
#pragma unroll
        for (int r = 0; r < 8; ++r)
            xg[(size_t)(mBase + r) * G4 + col] = acc[s][r] + bv;
    }
}

// ---------------- phase 2: persistent LSTM scan ----------------
// One launch runs all SEQ steps. 8 waves per block = 4 gates x 2 K-halves.
// The block's W_hg slice (256KB) lives in LDS for the whole sequence;
// h is re-staged into LDS each step via the Tensor Data Mover.
__global__ void __launch_bounds__(256, 1) k_lstm_persistent(
    const float*  __restrict__ xg,    // [SEQ, G4]
    const __bf16* __restrict__ whT,   // [G4, HID]
    __bf16* __restrict__ hbuf0,       // [HID]
    __bf16* __restrict__ hbuf1,       // [HID]
    float*  __restrict__ out,         // [SEQ, HID]
    unsigned* __restrict__ bar) {
    extern __shared__ char smem[];
    __bf16* wlds = (__bf16*)smem;
    __bf16* hsh  = (__bf16*)(smem + HSH_OFF);
    float (*gbuf)[4][16] = (float (*)[4][16])(smem + GBUF_OFF);

    const int tid  = threadIdx.x;
    const int lane = tid & 31;
    const int wid  = tid >> 5;        // 0..7
    const int gate = wid & 3;         // o, f, i, g
    const int half = wid >> 2;        // K half: 0 or 1
    const int jBase = (int)blockIdx.x * JB;
    const bool hi  = lane >= 16;
    const int  l15 = lane & 15;

    // ---- one-time: stage this block's W slice into LDS (64 rows, padded stride) ----
    for (int idx = tid; idx < 64 * (HID / 8); idx += 256) {
        const int row = idx / (HID / 8);       // 0..63 = gate*16 + col
        const int seg = idx % (HID / 8);       // 16B segment within row
        const int g   = row >> 4;
        const int r   = row & 15;
        const __bf16* src = whT + (size_t)(g * HID + jBase + r) * HID + seg * 8;
        *(uint4*)(wlds + (size_t)row * WROW + seg * 8) = *(const uint4*)src;
    }
    __syncthreads();

    const __bf16* bRowL =
        wlds + (size_t)(gate * 16 + l15) * WROW + half * (HID / 2) + (hi ? 16 : 0);

    float c_reg = 0.0f;

    for (int t = 0; t < SEQ; ++t) {
        const __bf16* hin  = (t & 1) ? hbuf1 : hbuf0;
        __bf16*       hout = (t & 1) ? hbuf0 : hbuf1;

        // ---- stage h[2048] (4KB) into LDS via TDM ----
#if __has_builtin(__builtin_amdgcn_tensor_load_to_lds)
        if (tid == 0) {
            const unsigned lds_base = (unsigned)(uintptr_t)&hsh[0];
            const unsigned long long ga = (unsigned long long)(uintptr_t)hin;
            v4u g0;
            g0.x = 1u;                                                 // count=1, user D#
            g0.y = lds_base;                                           // lds_addr
            g0.z = (unsigned)ga;                                       // global_addr[31:0]
            g0.w = (unsigned)((ga >> 32) & 0x1FFFFFFull) | (2u << 30); // addr[56:32], type=2
            v8i_ g1;
            g1[0] = (int)0x00010000;  // data_size=1 (2B), wg_mask=0
            g1[1] = (int)0x08000000;  // tensor_dim0 = 2048 (low16 @ bits 63:48)
            g1[2] = (int)0x00010000;  // tensor_dim1 = 1
            g1[3] = (int)0x08000000;  // tile_dim0 = 2048
            g1[4] = 1;                // tile_dim1 = 1
            g1[5] = 2048;             // tensor_dim0_stride
            g1[6] = 0;
            g1[7] = 0;
            v4i_ gz = {};
#if __clang_major__ >= 23
            v8i_ gz8 = {};
            __builtin_amdgcn_tensor_load_to_lds(g0, g1, gz, gz, gz8, 0);
#else
            __builtin_amdgcn_tensor_load_to_lds(g0, g1, gz, gz, 0);
#endif
        }
#if __has_builtin(__builtin_amdgcn_s_wait_tensorcnt)
        __builtin_amdgcn_s_wait_tensorcnt(0);
#else
        asm volatile("s_wait_tensorcnt 0x0" ::: "memory");
#endif
#else
        ((uint4*)hsh)[tid] = ((const uint4*)hin)[tid];
#endif
        __syncthreads();

        // ---- GEMV slice via broadcast-A WMMA, B resident in LDS ----
        v8f acc = {};
        const __bf16* hb = hsh + half * (HID / 2);
#pragma unroll 8
        for (int k0 = 0; k0 < HID / 2; k0 += 32) {
            ABf a;
            const int ka = k0 + (hi ? 16 : 0);
            a.q[0] = *(const uint4*)(hb + ka);        // ds_load_b128 (broadcast)
            a.q[1] = *(const uint4*)(hb + ka + 8);
            ABf b;
            b.q[0] = ((const uint4*)(bRowL + k0))[0]; // ds_load_b128, bank-conflict-free
            b.q[1] = ((const uint4*)(bRowL + k0))[1];
            acc = __builtin_amdgcn_wmma_f32_16x16x32_bf16(
                false, a.v, false, b.v, (short)0, acc, false, false);
        }
        if (!hi) gbuf[half][gate][lane] = acc[0];     // row M=0 of D
        __syncthreads();

        // ---- gate nonlinearities + state update ----
        if (tid < JB) {
            const int j = jBase + tid;
            const float* xgt = xg + (size_t)t * G4;
            const float o = sigmoidf_(xgt[j]           + gbuf[0][0][tid] + gbuf[1][0][tid]);
            const float f = sigmoidf_(xgt[HID + j]     + gbuf[0][1][tid] + gbuf[1][1][tid]);
            const float i = sigmoidf_(xgt[2 * HID + j] + gbuf[0][2][tid] + gbuf[1][2][tid]);
            const float g = tanhf    (xgt[3 * HID + j] + gbuf[0][3][tid] + gbuf[1][3][tid]);
            c_reg = f * c_reg + i * g;
            const float hn = o * tanhf(c_reg);
            out[(size_t)t * HID + j] = hn;
            hout[j] = f2bf(hn);
        }
        __syncthreads();

        // ---- device-wide step barrier (generation counter) ----
        if (tid == 0) {
            unsigned arr = __hip_atomic_fetch_add(&bar[0], 1u, __ATOMIC_ACQ_REL,
                                                  __HIP_MEMORY_SCOPE_AGENT);
            if (arr == NB - 1u) {
                __hip_atomic_store(&bar[0], 0u, __ATOMIC_RELAXED, __HIP_MEMORY_SCOPE_AGENT);
                __hip_atomic_store(&bar[1], (unsigned)(t + 1), __ATOMIC_RELEASE,
                                   __HIP_MEMORY_SCOPE_AGENT);
            }
        }
        if (lane == 0) {   // one spinner per wave -> each CU's cache sees the acquire
            while (__hip_atomic_load(&bar[1], __ATOMIC_ACQUIRE,
                                     __HIP_MEMORY_SCOPE_AGENT) < (unsigned)(t + 1))
                __builtin_amdgcn_s_sleep(2);
        }
        __syncthreads();
    }
}

// ---------------- launcher ----------------

extern "C" void kernel_launch(void* const* d_in, const int* in_sizes, int n_in,
                              void* d_out, int out_size, void* d_ws, size_t ws_size,
                              hipStream_t stream) {
    const float* x    = (const float*)d_in[0];   // [SEQ, INP]
    const float* W_xg = (const float*)d_in[1];   // [INP, G4]
    const float* W_hg = (const float*)d_in[2];   // [HID, G4]
    const float* b_g  = (const float*)d_in[3];   // [G4]
    float* out = (float*)d_out;                  // [SEQ, HID]

    char* ws = (char*)d_ws;
    size_t off = 0;
    float*  xg  = (float*)(ws + off);  off += (size_t)SEQ * G4  * sizeof(float);   // 256 MB
    __bf16* xb  = (__bf16*)(ws + off); off += (size_t)SEQ * INP * sizeof(__bf16);  // 32 MB
    __bf16* wxT = (__bf16*)(ws + off); off += (size_t)G4  * INP * sizeof(__bf16);  // 32 MB
    __bf16* whT = (__bf16*)(ws + off); off += (size_t)G4  * HID * sizeof(__bf16);  // 32 MB
    __bf16* h0  = (__bf16*)(ws + off); off += (size_t)HID * sizeof(__bf16);
    __bf16* h1  = (__bf16*)(ws + off); off += (size_t)HID * sizeof(__bf16);
    unsigned* bar = (unsigned*)(ws + off); off += 2 * sizeof(unsigned);
    (void)ws_size; (void)in_sizes; (void)n_in; (void)out_size;

    {
        const int n = SEQ * INP;
        k_cast_bf16<<<(n + 255) / 256, 256, 0, stream>>>(x, xb, n);
    }
    {
        dim3 grid(INP / 32, G4 / 32);
        k_tcast<<<grid, 256, 0, stream>>>(W_xg, wxT, INP, G4);
        k_tcast<<<grid, 256, 0, stream>>>(W_hg, whT, HID, G4);
    }
    k_init_state<<<(HID + 255) / 256, 256, 0, stream>>>(h0, h1, bar);

    {
        const int waves  = (SEQ / 16) * (G4 / 64);   // 65536
        const int blocks = waves / 8;                // 8192
        k_gemm_xw<<<blocks, 256, 0, stream>>>(xb, wxT, b_g, xg);
    }

    // Persistent scan: raise dynamic-LDS cap (host-side, capture-safe), then launch.
    (void)hipFuncSetAttribute((const void*)k_lstm_persistent,
                              hipFuncAttributeMaxDynamicSharedMemorySize, SMEM_BYTES);
    k_lstm_persistent<<<NB, 256, SMEM_BYTES, stream>>>(xg, whT, h0, h1, out, bar);
}
